// RKMeans_Tokenizer_76965813945018
// MI455X (gfx1250) — matrix operand order repaired and checked by reading, not compiled, not run on hardware
//
#include <hip/hip_runtime.h>
#include <hip/hip_bf16.h>

typedef __attribute__((ext_vector_type(16))) _Float16 v16h;
typedef __attribute__((ext_vector_type(8)))  float    v8f;

#define RK_D 384
#define RK_K 256
#define RK_L 3
#define RK_KC (RK_D / 32)          // 12 k-chunks per row
#define ROWS_PER_WAVE 16
#define WAVES_PER_BLOCK 8
#define ROWS_PER_BLOCK (ROWS_PER_WAVE * WAVES_PER_BLOCK)

// ---------------------------------------------------------------------------
// Prep: convert f32 codebooks -> f16 (WMMA B operands) and compute
// per-centroid squared norms. One block per centroid (L*K = 768 blocks).
// ---------------------------------------------------------------------------
__global__ void rkm_prep(const float* __restrict__ cb,
                         _Float16* __restrict__ cbh,
                         float* __restrict__ cnorm) {
    const int c   = blockIdx.x;         // centroid index in [0, L*K)
    const int tid = threadIdx.x;        // 128 threads = 4 waves
    const float*    src = cb  + (size_t)c * RK_D;
    _Float16*       dst = cbh + (size_t)c * RK_D;

    float ss = 0.0f;
    for (int j = tid; j < RK_D; j += 128) {
        float v = src[j];
        dst[j] = (_Float16)v;
        ss += v * v;
    }
    #pragma unroll
    for (int off = 16; off >= 1; off >>= 1) ss += __shfl_xor(ss, off, 32);

    __shared__ float red[4];
    const int lane = tid & 31, w = tid >> 5;
    if (lane == 0) red[w] = ss;
    __syncthreads();
    if (tid == 0) cnorm[c] = red[0] + red[1] + red[2] + red[3];
}

// ---------------------------------------------------------------------------
// Main: each wave owns 16 embedding rows, staged HBM->LDS via CDNA5 async
// loads, kept as an f32 residual tile in LDS across all 3 layers. Per layer,
// the wave's 16x384 A-matrix is converted to f16 WMMA fragments ONCE
// (12 x v16h = 96 VGPRs) and reused for all 16 centroid tiles, so the hot
// loop is just {B-load b128x2, v_wmma} pairs.
// ---------------------------------------------------------------------------
__global__ void __launch_bounds__(256, 1)
rkm_main(const float* __restrict__ emb,
         const float* __restrict__ cb,
         const _Float16* __restrict__ cbh,
         const float* __restrict__ cnorm,
         float* __restrict__ out, int N) {
    extern __shared__ char smem[];
    float* res   = (float*)smem;                                   // [8][16][384] f32
    int*   sid_s = (int*)(smem + (size_t)WAVES_PER_BLOCK * ROWS_PER_WAVE * RK_D * sizeof(float));

    const int tid  = threadIdx.x;
    const int lane = tid & 31;
    const int wid  = tid >> 5;
    const int hl   = lane & 15;   // half-lane (column / A-row id)
    const int hi   = lane >> 4;   // which 16-lane half
    const int rowBase = (blockIdx.x * WAVES_PER_BLOCK + wid) * ROWS_PER_WAVE;
    float* wres = res + (size_t)wid * ROWS_PER_WAVE * RK_D;

    // ---- stage the 16x384 f32 embedding tile HBM -> LDS with async copies
    //      (global_load_async_to_lds_b128: no VGPR round-trip, ASYNCcnt) ----
    for (int r = 0; r < ROWS_PER_WAVE; ++r) {
        int row = rowBase + r; if (row >= N) row = N - 1;   // clamp: dup work, no dup store
        const unsigned long long gbase =
            (unsigned long long)(emb + (size_t)row * RK_D);
        const unsigned ldsbase =
            (unsigned)(((wid * ROWS_PER_WAVE + r) * RK_D) * sizeof(float));
        #pragma unroll
        for (int j = 0; j < 3; ++j) {
            const unsigned      ldsoff = ldsbase + (unsigned)((lane + 32 * j) * 16);
            const unsigned long long ga = gbase + (unsigned long long)((lane + 32 * j) * 16);
            asm volatile("global_load_async_to_lds_b128 %0, %1, off"
                         :: "v"(ldsoff), "v"(ga) : "memory");
        }
    }
    asm volatile("s_wait_asynccnt 0" ::: "memory");
    __syncthreads();

    float* resOut = out + (size_t)N * RK_L;

    for (int layer = 0; layer < RK_L; ++layer) {
        // ---- 1. normalize rows in place: r /= max(||r||, eps) ----
        for (int r = 0; r < ROWS_PER_WAVE; ++r) {
            float4* prow = (float4*)(wres + (size_t)r * RK_D);
            float ss = 0.0f;
            #pragma unroll
            for (int j = 0; j < 3; ++j) {
                float4 v = prow[lane + 32 * j];
                ss += v.x * v.x + v.y * v.y + v.z * v.z + v.w * v.w;
            }
            #pragma unroll
            for (int off = 16; off >= 1; off >>= 1) ss += __shfl_xor(ss, off, 32);
            const float inv = 1.0f / fmaxf(sqrtf(ss), 1e-8f);
            #pragma unroll
            for (int j = 0; j < 3; ++j) {
                float4 v = prow[lane + 32 * j];
                v.x *= inv; v.y *= inv; v.z *= inv; v.w *= inv;
                prow[lane + 32 * j] = v;
            }
        }
        __syncthreads();   // A build reads rows cross-lane

        // ---- 2. build the layer's A fragments once (16x32 f16 layout) ----
        const float* arow = wres + (size_t)hl * RK_D;   // this lane's A row (M = hl)
        v16h afrag[RK_KC];                              // 12 x 8 VGPRs, reused 16x
        #pragma unroll
        for (int kc = 0; kc < RK_KC; ++kc) {
            const float* ap = arow + kc * 32;
            v16h a;
            #pragma unroll
            for (int j = 0; j < 8; ++j) {
                const int kb = hi * 16 + (j << 1);
                float2 p = *(const float2*)(ap + kb);
                a[2 * j]     = (_Float16)p.x;
                a[2 * j + 1] = (_Float16)p.y;
            }
            afrag[kc] = a;
        }

        // ---- 3. argmin over 256 centroids: hot loop = B-load + WMMA ----
        const _Float16* cbl = cbh   + (size_t)layer * RK_K * RK_D;
        const float*    cnl = cnorm + (size_t)layer * RK_K;

        float bestVal[8];
        int   bestIdx[8];
        #pragma unroll
        for (int v = 0; v < 8; ++v) { bestVal[v] = 3.4e38f; bestIdx[v] = 0; }

        for (int t = 0; t < RK_K / 16; ++t) {           // 16 centroid tiles
            if (t + 1 < RK_K / 16)                      // warm WGP$/L2 for next tile
                __builtin_prefetch(cbl + (size_t)((t + 1) * 16 + hl) * RK_D, 0, 1);

            // B fragments: column = centroid t*16+hl, 16 contiguous f16, 32B aligned
            const _Float16* bbase = cbl + (size_t)(t * 16 + hl) * RK_D + hi * 16;
            v8f acc = {};
            #pragma unroll
            for (int kc = 0; kc < RK_KC; ++kc) {
                v16h b = *(const v16h*)(bbase + kc * 32);
                acc = __builtin_amdgcn_wmma_f32_16x16x32_f16(
                    /*neg_a=*/false, afrag[kc], /*neg_b=*/false, b,
                    /*c_mod=*/(short)0, acc, /*reuse_a=*/false, /*reuse_b=*/false);
            }
            // score = ||c||^2 - 2*(q.c); q-norm irrelevant for argmin
            const float cn  = cnl[t * 16 + hl];
            const int   idx = t * 16 + hl;
            #pragma unroll
            for (int v = 0; v < 8; ++v) {
                const float s = fmaf(-2.0f, acc[v], cn);
                if (s < bestVal[v]) { bestVal[v] = s; bestIdx[v] = idx; }
            }
        }
        // cross-lane argmin inside each 16-lane half (C layout: lane = column)
        #pragma unroll
        for (int v = 0; v < 8; ++v) {
            float val = bestVal[v]; int idx = bestIdx[v];
            #pragma unroll
            for (int off = 8; off >= 1; off >>= 1) {
                const float ov = __shfl_xor(val, off, 16);
                const int   oi = __shfl_xor(idx, off, 16);
                if (ov < val || (ov == val && oi < idx)) { val = ov; idx = oi; }
            }
            bestVal[v] = val; bestIdx[v] = idx;
        }
        if (hl == 0) {       // lanes 0/16 hold rows 0..7 / 8..15 (VGPR v -> row)
            #pragma unroll
            for (int v = 0; v < 8; ++v) sid_s[wid * 16 + hi * 8 + v] = bestIdx[v];
        }
        __syncthreads();

        // ---- 4. emit sid (as float: int32+f32 tuple flattens to f32) ----
        if (lane < 16) {
            const int row = rowBase + lane;
            if (row < N)
                out[(size_t)row * RK_L + layer] = (float)sid_s[wid * 16 + lane];
        }
        // ---- 5. residual = normalized - centroid[sid]  (f32 codebook) ----
        const float* cbl32 = cb + (size_t)layer * RK_K * RK_D;
        for (int r = 0; r < ROWS_PER_WAVE; ++r) {
            const int sid = sid_s[wid * 16 + r];
            const float4* crow = (const float4*)(cbl32 + (size_t)sid * RK_D);
            float4*       prow = (float4*)(wres + (size_t)r * RK_D);
            #pragma unroll
            for (int j = 0; j < 3; ++j) {
                float4 v = prow[lane + 32 * j];
                const float4 c = crow[lane + 32 * j];
                v.x -= c.x; v.y -= c.y; v.z -= c.z; v.w -= c.w;
                prow[lane + 32 * j] = v;
            }
        }
        __syncthreads();
    }

    // ---- final residual store (single pass to HBM) ----
    for (int r = 0; r < ROWS_PER_WAVE; ++r) {
        const int row = rowBase + r;
        if (row < N) {
            float4*       dst  = (float4*)(resOut + (size_t)row * RK_D);
            const float4* srcp = (const float4*)(wres + (size_t)r * RK_D);
            #pragma unroll
            for (int j = 0; j < 3; ++j) dst[lane + 32 * j] = srcp[lane + 32 * j];
        }
    }
}

extern "C" void kernel_launch(void* const* d_in, const int* in_sizes, int n_in,
                              void* d_out, int out_size, void* d_ws, size_t ws_size,
                              hipStream_t stream) {
    const float* emb = (const float*)d_in[0];   // [N, 384] f32
    const float* cb  = (const float*)d_in[1];   // [3, 256, 384] f32
    const int N = in_sizes[0] / RK_D;

    _Float16* cbh   = (_Float16*)d_ws;                                   // 576 KB
    float*    cnorm = (float*)((char*)d_ws +
                               (size_t)RK_L * RK_K * RK_D * sizeof(_Float16));

    rkm_prep<<<RK_L * RK_K, 128, 0, stream>>>(cb, cbh, cnorm);

    const int blocks = (N + ROWS_PER_BLOCK - 1) / ROWS_PER_BLOCK;
    const size_t smem = (size_t)WAVES_PER_BLOCK * ROWS_PER_WAVE * RK_D * sizeof(float)
                      + (size_t)WAVES_PER_BLOCK * ROWS_PER_WAVE * sizeof(int); // 197120 B
    rkm_main<<<blocks, 256, smem, stream>>>(emb, cb, cbh, cnorm, (float*)d_out, N);
}
